// EdgeClassifier_44023414784046
// MI455X (gfx1250) — matrix operand order repaired and checked by reading, not compiled, not run on hardware
//
#include <hip/hip_runtime.h>
#include <stdint.h>

// ---------------------------------------------------------------------------
// EdgeClassifier: out[e] = w2 . relu(W1 @ [u, v, |u-v|, u*v] + b1) + b2
// bf16 WMMA (v_wmma_f32_16x16x32_bf16), 16 edges per wave-tile,
// w1 fragments resident in LDS, T tiles per wave.
// ---------------------------------------------------------------------------

typedef __attribute__((ext_vector_type(16))) __bf16 v16bf;
typedef __attribute__((ext_vector_type(8)))  float  v8f;

#define WAVES_PER_BLOCK 2           // 64 threads
#define TILES_PER_WAVE  16
#define TILE_EDGES      16
#define IN_DIM          256
#define NODE_D          64
#define FEAT_STRIDE     132         // 128 u32 payload + 4 pad (bank spread)
#define W1_FRAG_U32     8192        // 32 frags * 32 lanes * 8 u32 = 32 KB

// Native bf16 convert (hardware v_cvt_*bf16* path, RNE) instead of manual
// integer rounding — keeps the gather/pack stage off the integer ALU path.
__device__ __forceinline__ unsigned int pack_bf16x2(float a, float b) {
    union { __bf16 h[2]; unsigned int u; } c;
    c.h[0] = (__bf16)a;
    c.h[1] = (__bf16)b;
    return c.u;
}

__device__ __forceinline__ uint4 pack8(float4 a, float4 b) {
    return make_uint4(pack_bf16x2(a.x, a.y), pack_bf16x2(a.z, a.w),
                      pack_bf16x2(b.x, b.y), pack_bf16x2(b.z, b.w));
}

// ---------------------------------------------------------------------------
// Kernel 1: pack w1 [64][256] f32 -> WMMA B fragments (bf16) in workspace.
// Fragment (kc,nt) covers K = 32*kc..+31, N = 16*nt..+15.
// B layout (CDNA5 ISA 7.12.2): lane encodes N: n_local = lane&15
// reg j (0..7), half h encodes K: k_local = 16*(j>>2) + 8*(lane>>4) + 2*(j&3) + h.
// Plane-split storage (bank spread for LDS reads):
//   u32 index = frag*256 + (j>>2)*128 + lane*4 + (j&3)
// ---------------------------------------------------------------------------
__global__ void pack_w1_kernel(const float* __restrict__ w1,
                               unsigned int* __restrict__ wsB) {
    int idx = blockIdx.x * blockDim.x + threadIdx.x;   // 8192 total
    if (idx >= W1_FRAG_U32) return;
    int reg  = idx & 7;
    int lane = (idx >> 3) & 31;
    int frag = idx >> 8;            // kc*4 + nt
    int kc = frag >> 2;
    int gk = kc * 32 + ((reg >> 2) << 4) + ((lane >> 4) << 3) + 2 * (reg & 3);
    int n0 = (frag & 3) * 16 + (lane & 15);
    float a = w1[n0 * IN_DIM + gk];
    float b = w1[n0 * IN_DIM + gk + 1];
    int dst = frag * 256 + (reg >> 2) * 128 + lane * 4 + (reg & 3);
    wsB[dst] = pack_bf16x2(a, b);
}

// ---------------------------------------------------------------------------
// Kernel 2: main edge MLP.
// ---------------------------------------------------------------------------
__global__ void __launch_bounds__(WAVES_PER_BLOCK * 32)
edge_mlp_kernel(const float* __restrict__ h,
                const long long* __restrict__ ep,     // [2][E] int64
                const float* __restrict__ b1,
                const float* __restrict__ w2,
                const float* __restrict__ b2,
                const uint4* __restrict__ wsB,        // packed w1 fragments
                float* __restrict__ out,
                int nEdges, int nTiles) {
    __shared__ unsigned int w1Lds[W1_FRAG_U32];                          // 32 KB
    __shared__ unsigned int feat[WAVES_PER_BLOCK][TILE_EDGES * FEAT_STRIDE];

    const int lane = threadIdx.x & 31;
    const int wave = threadIdx.x >> 5;

    // ---- stage packed w1 into LDS (once per block) ------------------------
    {
        uint4* dst = (uint4*)w1Lds;
        const int nt4 = WAVES_PER_BLOCK * 32;
#pragma unroll
        for (int i = 0; i < (W1_FRAG_U32 / 4) / (WAVES_PER_BLOCK * 32); ++i)
            dst[threadIdx.x + i * nt4] = wsB[threadIdx.x + i * nt4];
    }
    __syncthreads();

    // ---- per-lane bias / final-layer weights (loop-invariant) -------------
    const int nl = lane & 15;
    float b1v[4], w2v[4];
#pragma unroll
    for (int nt = 0; nt < 4; ++nt) {
        b1v[nt] = b1[nt * 16 + nl];
        w2v[nt] = w2[nt * 16 + nl];
    }
    const float bias2 = b2[0];

    const int el = lane >> 1;       // edge-in-tile for gather (2 lanes/edge)
    const int hf = lane & 1;        // which 32-dim half
    unsigned int* frow = &feat[wave][el * FEAT_STRIDE + hf * 16];
    const unsigned int* fbase = &feat[wave][0];
    const int arow  = nl * FEAT_STRIDE;   // A-frag row base (M = lane&15)
    const int koff4 = (lane >> 4) * 4;    // lanes>=16 hold K+8 (=4 u32)
    const uint4* w1Lds4 = (const uint4*)w1Lds;

    const int tileBase = blockIdx.x * (WAVES_PER_BLOCK * TILES_PER_WAVE)
                       + wave * TILES_PER_WAVE;

    for (int t = 0; t < TILES_PER_WAVE; ++t) {
        const int tile = tileBase + t;
        const int tcl  = (tile < nTiles) ? tile : (nTiles - 1);  // uniform EXEC
        const long long e0 = (long long)tcl * TILE_EDGES;

        // prefetch next tile's edge indices into cache
        if (t + 1 < TILES_PER_WAVE) {
            int tn = tileBase + t + 1;
            tn = (tn < nTiles) ? tn : (nTiles - 1);
            const long long en = (long long)tn * TILE_EDGES;
            __builtin_prefetch(&ep[en + el], 0, 0);
            __builtin_prefetch(&ep[(long long)nEdges + en + el], 0, 0);
        }

        // ---- gather endpoints, build bf16 feature tile in LDS -------------
        {
            const int su = (int)ep[e0 + el];
            const int sv = (int)ep[(long long)nEdges + e0 + el];
            const float4* pu = (const float4*)(h + (long long)su * NODE_D + hf * 32);
            const float4* pv = (const float4*)(h + (long long)sv * NODE_D + hf * 32);
#pragma unroll
            for (int j = 0; j < 8; j += 2) {
                float4 ua = pu[j], ub = pu[j + 1];
                float4 va = pv[j], vb = pv[j + 1];
                float4 da, db, ma, mb;
                da.x = fabsf(ua.x - va.x); da.y = fabsf(ua.y - va.y);
                da.z = fabsf(ua.z - va.z); da.w = fabsf(ua.w - va.w);
                db.x = fabsf(ub.x - vb.x); db.y = fabsf(ub.y - vb.y);
                db.z = fabsf(ub.z - vb.z); db.w = fabsf(ub.w - vb.w);
                ma.x = ua.x * va.x; ma.y = ua.y * va.y;
                ma.z = ua.z * va.z; ma.w = ua.w * va.w;
                mb.x = ub.x * vb.x; mb.y = ub.y * vb.y;
                mb.z = ub.z * vb.z; mb.w = ub.w * vb.w;
                // segments: 0=u, 1=v, 2=|u-v|, 3=u*v ; u32 col = seg*32 + hf*16 + j*2
                ((uint4*)(frow + 0 * 32 + j * 2))[0] = pack8(ua, ub);
                ((uint4*)(frow + 1 * 32 + j * 2))[0] = pack8(va, vb);
                ((uint4*)(frow + 2 * 32 + j * 2))[0] = pack8(da, db);
                ((uint4*)(frow + 3 * 32 + j * 2))[0] = pack8(ma, mb);
            }
        }
        // No block barrier needed: each wave reads only its own feat region,
        // and per-wave LDS ops complete in order (DScnt).

        // ---- WMMA: D[16x16] += A[16x32] x B[32x16] over 8 K-chunks --------
        union AB { v16bf v; uint4 u[2]; };
        v8f acc[4] = {};
#pragma unroll
        for (int kc = 0; kc < 8; ++kc) {
            AB a;
            const uint4* ap = (const uint4*)(fbase + arow + kc * 16 + koff4);
            a.u[0] = ap[0];       // K .. K+7
            a.u[1] = ap[2];       // K+16 .. K+23   (+8 u32)
#pragma unroll
            for (int nt = 0; nt < 4; ++nt) {
                AB bf;
                const int frag = kc * 4 + nt;
                bf.u[0] = w1Lds4[frag * 64 + lane];        // plane 0 (regs 0-3)
                bf.u[1] = w1Lds4[frag * 64 + 32 + lane];   // plane 1 (regs 4-7)
                acc[nt] = __builtin_amdgcn_wmma_f32_16x16x32_bf16(
                    false, a.v, false, bf.v, (short)0, acc[nt], false, false);
            }
        }

        // ---- epilogue: relu + w2 dot, reduce over 16-lane N groups --------
        float s[8];
#pragma unroll
        for (int r = 0; r < 8; ++r) {
            float tacc = 0.0f;
#pragma unroll
            for (int nt = 0; nt < 4; ++nt) {
                float hv = acc[nt][r] + b1v[nt];
                hv = fmaxf(hv, 0.0f);
                tacc += hv * w2v[nt];
            }
            s[r] = tacc;
        }
#pragma unroll
        for (int r = 0; r < 8; ++r) {
#pragma unroll
            for (int m = 1; m < 16; m <<= 1)
                s[r] += __shfl_xor(s[r], m, 32);
        }

        // C/D layout: VGPR r -> M = r + 8*(lane>=16); lane%16 -> N (reduced)
        if (nl == 0 && tile < nTiles) {
            float4 o0, o1;
            o0.x = s[0] + bias2; o0.y = s[1] + bias2;
            o0.z = s[2] + bias2; o0.w = s[3] + bias2;
            o1.x = s[4] + bias2; o1.y = s[5] + bias2;
            o1.z = s[6] + bias2; o1.w = s[7] + bias2;
            float4* op = (float4*)(out + e0 + (lane >> 4) * 8);
            op[0] = o0;
            op[1] = o1;
        }
    }
}

// ---------------------------------------------------------------------------
extern "C" void kernel_launch(void* const* d_in, const int* in_sizes, int n_in,
                              void* d_out, int out_size, void* d_ws, size_t ws_size,
                              hipStream_t stream) {
    const float*     h_  = (const float*)d_in[0];
    const long long* ep  = (const long long*)d_in[1];
    const float*     w1  = (const float*)d_in[2];
    const float*     b1  = (const float*)d_in[3];
    const float*     w2  = (const float*)d_in[4];
    const float*     b2  = (const float*)d_in[5];
    unsigned int*    wsB = (unsigned int*)d_ws;
    float*           out = (float*)d_out;

    const int nEdges = in_sizes[1] / 2;          // 1,000,000
    const int nTiles = nEdges / TILE_EDGES;      // 62,500 (E % 16 == 0)
    const int tilesPerBlock = WAVES_PER_BLOCK * TILES_PER_WAVE;
    const int nBlocks = (nTiles + tilesPerBlock - 1) / tilesPerBlock;

    pack_w1_kernel<<<(W1_FRAG_U32 + 255) / 256, 256, 0, stream>>>(w1, wsB);
    edge_mlp_kernel<<<nBlocks, WAVES_PER_BLOCK * 32, 0, stream>>>(
        h_, ep, b1, w2, b2, (const uint4*)d_ws, out, nEdges, nTiles);
}
